// Mamba_65180423685574
// MI455X (gfx1250) — matrix operand order repaired and checked by reading, not compiled, not run on hardware
//
#include <hip/hip_runtime.h>
#include <hip/hip_bf16.h>
#include <stdint.h>

#define D_MODEL 1024
#define D_STATE 16
#define D_CONV  4
#define D_INNER 2048
#define BATCH   2
#define SEQLEN  2048
#define MROWS   (BATCH*SEQLEN)   /* 4096 rows through both GEMMs */

typedef __bf16 bf16;
typedef __attribute__((ext_vector_type(8)))  __bf16 v8bf;
typedef __attribute__((ext_vector_type(16))) __bf16 v16bf;
typedef __attribute__((ext_vector_type(8)))  float  v8f;

/* ---- gfx1250 async global->LDS path (guarded; falls back to plain copies) ---- */
#if defined(__has_builtin)
#if __has_builtin(__builtin_amdgcn_global_load_async_to_lds_b128) && \
    __has_builtin(__builtin_amdgcn_s_wait_asynccnt)
#define HAVE_ASYNC_LDS 1
#endif
#endif

#if defined(HAVE_ASYNC_LDS)
/* Builtin signature (from hipcc diagnostic): parameters are pointers to
 * 'int __attribute__((vector_size(16)))' in the global (AS1, printed
 * "__device__") and LDS (AS3) address spaces.                              */
typedef int v4i_ __attribute__((vector_size(16)));
typedef __attribute__((address_space(1))) v4i_ as1_v4i;
typedef __attribute__((address_space(3))) v4i_ as3_v4i;

static __device__ __forceinline__ as1_v4i* to_gbl(const void* p) {
  return (as1_v4i*)(uintptr_t)p;
}
/* LDS flat-aperture addresses carry the LDS offset in the low 32 bits
 * (CDNA5 ISA 10.2); AS3 pointers are 32-bit. */
static __device__ __forceinline__ as3_v4i* to_lds(void* p) {
  return (as3_v4i*)(uint32_t)(uintptr_t)p;
}
#endif

/* ---------------- fp32 -> bf16 conversion kernels ---------------- */

__global__ __launch_bounds__(256)
void cvt_bf16(const float* __restrict__ in, bf16* __restrict__ out, int n) {
  int i = blockIdx.x * 256 + threadIdx.x;
  if (i < n) out[i] = (bf16)in[i];
}

/* W: K x N row-major  ->  Wt: N x K row-major (bf16) */
__global__ __launch_bounds__(256)
void cvt_bf16_transpose(const float* __restrict__ W, bf16* __restrict__ Wt,
                        int K, int N) {
  int i = blockIdx.x * 256 + threadIdx.x;
  if (i < K * N) {
    int n = i / K;
    int k = i - n * K;
    Wt[i] = (bf16)W[(size_t)k * N + n];
  }
}

/* ---------------- bf16 WMMA GEMM: C(MxN,f32) = A(MxK) * Bt(NxK)^T ----------------
 * Block = 256 threads = 8 waves, 2(M) x 4(N); each wave owns a 16x64 strip as 4
 * v_wmma_f32_16x16x32_bf16 accumulators. The k-loop is a 2-deep software pipeline
 * manually unrolled x2 into ping/pong register buffer sets, so tile t+1's 6
 * loads issue before tile t's 4 WMMAs and NO rotation v_movs are needed.
 * Requires K/32 even (K = 1024 and 2048 here).
 * Fragment layout used here, contiguous k, not the ISA table order (16-bit A 16x32 / B 32x16):
 *   A: lane<16 holds K 0..15 of row M=lane;         lane>=16 holds K 16..31
 *   B: lane<16 holds K 0..15 of col N=lane;         lane>=16 holds K 16..31        */
__global__ __launch_bounds__(256)
void gemm_bf16_wmma(const bf16* __restrict__ A,   // M x K
                    const bf16* __restrict__ Bt,  // N x K (transposed weights)
                    float* __restrict__ C,        // M x N
                    int M, int N, int K) {
  const int lane = threadIdx.x & 31;
  const int wave = threadIdx.x >> 5;
  const int m0 = blockIdx.y * 32  + (wave >> 2) * 16;
  const int n0 = blockIdx.x * 256 + (wave & 3) * 64;
  const int lr = lane & 15;
  const int ka = (lane < 16) ? 0 : 16;   // A-fragment K phase
  const int kb = (lane < 16) ? 0 : 16;   // B-fragment K phase

  const bf16* pa = A + (size_t)(m0 + lr) * K;
  const bf16* pb[4];
#pragma unroll
  for (int j = 0; j < 4; ++j)
    pb[j] = Bt + (size_t)(n0 + j * 16 + lr) * K;

  v8f zero = {0.f, 0.f, 0.f, 0.f, 0.f, 0.f, 0.f, 0.f};
  v8f acc[4];
#pragma unroll
  for (int j = 0; j < 4; ++j) acc[j] = zero;

  /* fragment loader / consumer for one 32-wide k-tile */
  auto load_tile = [&](int kk, v8bf& lo, v8bf& hi, v16bf (&bfr)[4]) {
    lo = *(const v8bf*)(pa + kk + ka);
    hi = *(const v8bf*)(pa + kk + ka + 8);
#pragma unroll
    for (int j = 0; j < 4; ++j)
      bfr[j] = *(const v16bf*)(pb[j] + kk + kb);
  };
  auto do_wmma = [&](const v8bf& lo, const v8bf& hi, const v16bf (&bfr)[4]) {
    v16bf afrag = __builtin_shufflevector(lo, hi,
        0,1,2,3,4,5,6,7,8,9,10,11,12,13,14,15);
#pragma unroll
    for (int j = 0; j < 4; ++j)
      acc[j] = __builtin_amdgcn_wmma_f32_16x16x32_bf16(
          false, afrag, false, bfr[j], (short)0, acc[j], false, false);
  };

  v8bf lo0, hi0, lo1, hi1;
  v16bf b0[4], b1[4];

  load_tile(0, lo0, hi0, b0);                    /* prologue: tile 0 -> ping */
  int kk = 0;
  for (; kk + 64 < K; kk += 64) {
    load_tile(kk + 32, lo1, hi1, b1);            /* tile t+1 -> pong */
    __builtin_prefetch(pa + kk + 160, 0, 3);
    do_wmma(lo0, hi0, b0);                       /* consume ping (tile t)   */
    load_tile(kk + 64, lo0, hi0, b0);            /* tile t+2 -> ping */
    do_wmma(lo1, hi1, b1);                       /* consume pong (tile t+1) */
  }
  /* epilogue: ping holds tile K/32-2; fetch and finish last tile */
  load_tile(kk + 32, lo1, hi1, b1);
  do_wmma(lo0, hi0, b0);
  do_wmma(lo1, hi1, b1);

  /* C/D layout: VGPR i -> row m0 + i + (lane<16 ? 0 : 8), col = n-tile + (lane&15) */
  const int rb = m0 + ((lane < 16) ? 0 : 8);
#pragma unroll
  for (int j = 0; j < 4; ++j) {
    const int col = n0 + j * 16 + lr;
#pragma unroll
    for (int i = 0; i < 8; ++i)
      C[(size_t)(rb + i) * N + col] = acc[j][i];
  }
}

/* ---------------- fused depthwise conv + SiLU + proj + softplus(delta) ----------
 * One block per (b,t). 256 threads cover 2048 channels (8 each).                 */
__global__ __launch_bounds__(256)
void conv_proj_delta(const float* __restrict__ xz,     // (B*L) x 4096; u half = [:,0:2048]
                     const float* __restrict__ conv_w, // 2048 x 4
                     const float* __restrict__ conv_b, // 2048
                     const float* __restrict__ W_x,    // 2048 x 32
                     const float* __restrict__ W_dt,   // 16 x 2048
                     const float* __restrict__ b_dt,   // 2048
                     float* __restrict__ u_out,        // (B*L) x 2048
                     float* __restrict__ delta_out,    // (B*L) x 2048
                     float* __restrict__ Bm_out)       // (B*L) x 16
{
  __shared__ float u_sh[D_INNER];
  __shared__ float proj_sh[2 * D_STATE];
  const int bt = blockIdx.x;
  const int b  = bt / SEQLEN;
  const int t  = bt - b * SEQLEN;

  /* causal depthwise conv (left pad 3) + SiLU */
#pragma unroll
  for (int i = 0; i < 8; ++i) {
    const int d = threadIdx.x + 256 * i;
    float acc = conv_b[d];
#pragma unroll
    for (int k = 0; k < D_CONV; ++k) {
      const int tt = t - (D_CONV - 1) + k;
      if (tt >= 0)
        acc += xz[(size_t)(b * SEQLEN + tt) * (2 * D_INNER) + d] * conv_w[d * D_CONV + k];
    }
    const float s = acc / (1.0f + __expf(-acc));
    u_sh[d] = s;
    u_out[(size_t)bt * D_INNER + d] = s;
  }
  __syncthreads();

  /* proj = u @ W_x : 32 outputs, 4 per wave, wave32 shuffle reduction */
  const int wave = threadIdx.x >> 5, lane = threadIdx.x & 31;
  const int nb = wave * 4;
  float p0 = 0.f, p1 = 0.f, p2 = 0.f, p3 = 0.f;
  for (int d = lane; d < D_INNER; d += 32) {
    const float uv = u_sh[d];
    const float* wr = W_x + (size_t)d * (2 * D_STATE) + nb;
    p0 += uv * wr[0]; p1 += uv * wr[1]; p2 += uv * wr[2]; p3 += uv * wr[3];
  }
#pragma unroll
  for (int off = 16; off; off >>= 1) {
    p0 += __shfl_down(p0, off, 32);
    p1 += __shfl_down(p1, off, 32);
    p2 += __shfl_down(p2, off, 32);
    p3 += __shfl_down(p3, off, 32);
  }
  if (lane == 0) {
    proj_sh[nb + 0] = p0; proj_sh[nb + 1] = p1;
    proj_sh[nb + 2] = p2; proj_sh[nb + 3] = p3;
  }
  __syncthreads();

  if (threadIdx.x < D_STATE)
    Bm_out[(size_t)bt * D_STATE + threadIdx.x] = proj_sh[D_STATE + threadIdx.x];

  /* delta = softplus(delta_in @ W_dt + b_dt) */
#pragma unroll
  for (int i = 0; i < 8; ++i) {
    const int d = threadIdx.x + 256 * i;
    float acc = b_dt[d];
#pragma unroll
    for (int n = 0; n < D_STATE; ++n)
      acc += proj_sh[n] * W_dt[n * D_INNER + d];
    delta_out[(size_t)bt * D_INNER + d] = (acc > 20.f) ? acc : log1pf(__expf(acc));
  }
}

/* ---------------- sequential selective scan ----------------
 * Thread per (b,d), h[16] in VGPRs. All 256 threads of a block share the same b,
 * so B_t is staged per 128-step tile into double-buffered LDS with gfx1250
 * async global->LDS copies (fallback: plain copies); per-step reads become
 * ds_load_b128 broadcasts and the next tile's staging overlaps the exp chain.   */
#define SCAN_TILE 128

__global__ __launch_bounds__(256)
void scan_kernel(const float* __restrict__ delta, const float* __restrict__ u,
                 const float* __restrict__ Bm,    const float* __restrict__ A_log,
                 float* __restrict__ ys)
{
  __shared__ float Bsh[2][SCAN_TILE * D_STATE];   /* 2 x 8 KB */

  const int gid = blockIdx.x * 256 + threadIdx.x;   // 0 .. B*D_INNER-1
  const int b = gid / D_INNER;
  const int d = gid - b * D_INNER;

  float A[D_STATE], h[D_STATE];
#pragma unroll
  for (int n = 0; n < D_STATE; ++n) {
    A[n] = -__expf(A_log[d * D_STATE + n]);
    h[n] = 0.f;
  }

  const float* dp = delta + (size_t)b * SEQLEN * D_INNER + d;
  const float* up = u     + (size_t)b * SEQLEN * D_INNER + d;
  const float* bp = Bm    + (size_t)b * SEQLEN * D_STATE;
  float*       yp = ys    + (size_t)b * SEQLEN * D_INNER + d;

  /* stage one tile of B_t (SCAN_TILE*16 floats = 512 float4) into LDS buffer */
  auto stage = [&](int buf, int t0) {
    const float* src = bp + (size_t)t0 * D_STATE;
    for (int i = threadIdx.x; i < SCAN_TILE * D_STATE / 4; i += 256) {
#if defined(HAVE_ASYNC_LDS)
      __builtin_amdgcn_global_load_async_to_lds_b128(
          to_gbl(src + i * 4), to_lds(&Bsh[buf][i * 4]), 0, 0);
#else
      *(float4*)&Bsh[buf][i * 4] = *(const float4*)(src + i * 4);
#endif
    }
  };

  stage(0, 0);
#if defined(HAVE_ASYNC_LDS)
  __builtin_amdgcn_s_wait_asynccnt(0);
#endif
  __syncthreads();

  int cur = 0;
  for (int t0 = 0; t0 < SEQLEN; t0 += SCAN_TILE) {
    if (t0 + SCAN_TILE < SEQLEN)
      stage(cur ^ 1, t0 + SCAN_TILE);          /* async overlap with compute */

    for (int tt = 0; tt < SCAN_TILE; ++tt) {
      const int t = t0 + tt;
      const float dt = dp[(size_t)t * D_INNER];
      const float ut = up[(size_t)t * D_INNER];
      const float du = dt * ut;
      const float4* bq = (const float4*)&Bsh[cur][tt * D_STATE];
      const float4 B0 = bq[0], B1 = bq[1], B2 = bq[2], B3 = bq[3];
      float Bv[D_STATE] = {B0.x,B0.y,B0.z,B0.w, B1.x,B1.y,B1.z,B1.w,
                           B2.x,B2.y,B2.z,B2.w, B3.x,B3.y,B3.z,B3.w};
      float y = 0.f;
#pragma unroll
      for (int n = 0; n < D_STATE; ++n) {
        const float hn = __expf(dt * A[n]) * h[n] + du * Bv[n];
        h[n] = hn;
        y += hn * Bv[n];
      }
      yp[(size_t)t * D_INNER] = y;
    }

#if defined(HAVE_ASYNC_LDS)
    __builtin_amdgcn_s_wait_asynccnt(0);
#endif
    __syncthreads();          /* staged tile visible; old buffer free for reuse */
    cur ^= 1;
  }
}

/* ---------------- y = (ys + D*u) * silu(z), cast to bf16 for the out-GEMM ------ */
__global__ __launch_bounds__(256)
void gate_cvt(const float* __restrict__ ys, const float* __restrict__ u,
              const float* __restrict__ xz, const float* __restrict__ Dp,
              bf16* __restrict__ yb)
{
  const int i = blockIdx.x * 256 + threadIdx.x;   // over MROWS*D_INNER
  const int row = i / D_INNER;
  const int d   = i - row * D_INNER;
  const float yv = ys[i] + Dp[d] * u[i];
  const float zv = xz[(size_t)row * (2 * D_INNER) + D_INNER + d];
  const float g  = zv / (1.0f + __expf(-zv));
  yb[i] = (bf16)(yv * g);
}

/* ---------------- host-side orchestration ---------------- */

extern "C" void kernel_launch(void* const* d_in, const int* in_sizes, int n_in,
                              void* d_out, int out_size, void* d_ws, size_t ws_size,
                              hipStream_t stream) {
  const float* x      = (const float*)d_in[0];
  const float* W_in   = (const float*)d_in[1];
  const float* conv_w = (const float*)d_in[2];
  const float* conv_b = (const float*)d_in[3];
  const float* W_x    = (const float*)d_in[4];
  const float* W_dt   = (const float*)d_in[5];
  const float* b_dt   = (const float*)d_in[6];
  const float* A_log  = (const float*)d_in[7];
  const float* D_par  = (const float*)d_in[8];
  const float* W_out  = (const float*)d_in[9];
  float* out = (float*)d_out;

  uint8_t* p = (uint8_t*)d_ws;
  auto carve = [&](size_t bytes) -> void* {
    void* r = (void*)p;
    p += (bytes + 255) & ~(size_t)255;
    return r;
  };

  bf16*  xb   = (bf16*) carve((size_t)MROWS * D_MODEL * 2);           // x in bf16
  bf16*  wint = (bf16*) carve((size_t)(2*D_INNER) * D_MODEL * 2);     // W_in^T (4096 x 1024)
  bf16*  wott = (bf16*) carve((size_t)D_MODEL * D_INNER * 2);         // W_out^T (1024 x 2048)
  float* xz   = (float*)carve((size_t)MROWS * (2*D_INNER) * 4);       // x @ W_in
  float* uu   = (float*)carve((size_t)MROWS * D_INNER * 4);           // silu(conv(u))
  float* dl   = (float*)carve((size_t)MROWS * D_INNER * 4);           // delta
  float* Bmw  = (float*)carve((size_t)MROWS * D_STATE * 4);           // B_t
  float* ysb  = (float*)carve((size_t)MROWS * D_INNER * 4);           // scan output
  bf16*  yb   = (bf16*) carve((size_t)MROWS * D_INNER * 2);           // gated y in bf16
  (void)ws_size; (void)in_sizes; (void)n_in; (void)out_size;

  /* conversions */
  cvt_bf16<<<(MROWS * D_MODEL + 255) / 256, 256, 0, stream>>>(x, xb, MROWS * D_MODEL);
  cvt_bf16_transpose<<<((2*D_INNER) * D_MODEL + 255) / 256, 256, 0, stream>>>(
      W_in, wint, D_MODEL, 2 * D_INNER);
  cvt_bf16_transpose<<<(D_MODEL * D_INNER + 255) / 256, 256, 0, stream>>>(
      W_out, wott, D_INNER, D_MODEL);

  /* GEMM1: xz = x @ W_in   (4096 x 4096 x 1024) */
  gemm_bf16_wmma<<<dim3((2*D_INNER) / 256, MROWS / 32), 256, 0, stream>>>(
      xb, wint, xz, MROWS, 2 * D_INNER, D_MODEL);

  /* conv + silu + proj + delta */
  conv_proj_delta<<<MROWS, 256, 0, stream>>>(
      xz, conv_w, conv_b, W_x, W_dt, b_dt, uu, dl, Bmw);

  /* sequential selective scan */
  scan_kernel<<<(BATCH * D_INNER) / 256, 256, 0, stream>>>(dl, uu, Bmw, A_log, ysb);

  /* gate + bf16 cast */
  gate_cvt<<<(MROWS * D_INNER) / 256, 256, 0, stream>>>(ysb, uu, xz, D_par, yb);

  /* GEMM2: out = y @ W_out  (4096 x 1024 x 2048) */
  gemm_bf16_wmma<<<dim3(D_MODEL / 256, MROWS / 32), 256, 0, stream>>>(
      yb, wott, out, MROWS, D_MODEL, D_INNER);
}
